// UberCRSN_35656818492169
// MI455X (gfx1250) — matrix operand order repaired and checked by reading, not compiled, not run on hardware
//
#include <hip/hip_runtime.h>
#include <hip/hip_bf16.h>

// ---------------------------------------------------------------------------
// UberCRSN forward for MI455X (gfx1250, wave32, WMMA + Tensor Data Mover).
// All GEMM operands live as f16 in global memory; tiles are DMA'd into LDS by
// the TDM (tensor_load_to_lds, TENSORcnt), WMMA f32_16x16x32_f16 consumes
// them, f32 epilogue handles alpha/beta/bias and optional f16 C copy.
// ---------------------------------------------------------------------------

#define EPS_    1e-6f
#define THRESH_ 0.9999f
#define N_TOK_  2048
#define DIM_    512
#define VOCAB_  32000
#define DEPTH_  8
#define STACK_  32

typedef __attribute__((ext_vector_type(16))) _Float16 v16h;
typedef __attribute__((ext_vector_type(8)))  float    v8f;
typedef __attribute__((ext_vector_type(4)))  unsigned uint32x4;
typedef __attribute__((ext_vector_type(8)))  int      int32x8;
typedef __attribute__((ext_vector_type(4)))  int      int32x4;

#define BM 128
#define BN 64
#define BK 32
#define LDA (BK + 8)              // LDS row stride in halves (80B: bank-friendly)
#define SMEM_BYTES ((BM + BN) * LDA * 2)

#if __has_builtin(__builtin_amdgcn_tensor_load_to_lds) && \
    __has_builtin(__builtin_amdgcn_s_wait_tensorcnt)
#define USE_TDM 1
#else
#define USE_TDM 0
#endif

#if USE_TDM
// ---------------------------------------------------------------------------
// Issue a 2D TDM tile load: tile_rows x BK halves from row-major
// [rows_total, K] tensor at gaddr (tile start), into LDS at lds_byte_off with
// 4-DWORD padding inserted every 16 DWORDs (row stride 40 halves == LDA).
// D# layout per CDNA5 ISA §8.3 (group0) / §8.4 (group1).
// Toolchain note: this clang exposes the 6-arg builtin
//   (uint32x4 g0, int32x8 g1, int32x4 g2, int32x4 g3, int32x8 g4, i32 cpol)
// ---------------------------------------------------------------------------
__device__ __forceinline__ void tdm_load_tile(const _Float16* gaddr,
                                              unsigned lds_byte_off,
                                              int K, int rows_total,
                                              int tile_rows)
{
    const unsigned long long ga = (unsigned long long)gaddr;
    uint32x4 g0;
    g0[0] = 1u;                                        // count=1, user D#
    g0[1] = lds_byte_off;                              // lds_addr
    g0[2] = (unsigned)(ga & 0xFFFFFFFFull);            // global_addr[31:0]
    g0[3] = ((unsigned)(ga >> 32) & 0x1FFFFFFu)        // global_addr[56:32]
          | (2u << 30);                                // type=2 ("image")
    int32x8 g1;
    g1[0] = (1 << 16)                                  // data_size = 2 bytes
          | (1 << 20)                                  // pad_enable
          | (3 << 22)                                  // pad_interval: 16 DWORDs
          | (3 << 25);                                 // pad_amount: 4 DWORDs
    g1[1] = (K & 0xFFFF) << 16;                        // tensor_dim0[15:0]
    g1[2] = ((K >> 16) & 0xFFFF)                       // tensor_dim0[31:16]
          | ((rows_total & 0xFFFF) << 16);             // tensor_dim1[15:0]
    g1[3] = ((rows_total >> 16) & 0xFFFF)              // tensor_dim1[31:16]
          | (BK << 16);                                // tile_dim0 = 32
    g1[4] = tile_rows;                                 // tile_dim1 (tile_dim2=0)
    g1[5] = K;                                         // tensor_dim0_stride[31:0]
    g1[6] = 0;                                         // stride hi / dim1_stride
    g1[7] = 0;
    const int32x4 gz4 = {0, 0, 0, 0};                  // groups 2/3 unused (2D)
    const int32x8 gz8 = {0, 0, 0, 0, 0, 0, 0, 0};
    __builtin_amdgcn_tensor_load_to_lds(g0, g1, gz4, gz4, gz8, 0);
}
#endif

// ---------------------------------------------------------------------------
// NT GEMM, all-f16 operands: C[n,o] = beta*C + alpha*sum_k Ah[n,k]*Bh[o,k]
//                                     (+ bias[o]); optional f16 copy of C.
// Requires N%128==0, O%64==0, K%32==0 (true for every call here).
// 256 threads = 8 waves; tile 128x64; wave = 64x16 (4 stacked 16x16 WMMAs).
// ---------------------------------------------------------------------------
__global__ __launch_bounds__(256)
void k_gemm_hh(const _Float16* __restrict__ Ah, const _Float16* __restrict__ Bh,
               float* __restrict__ C, _Float16* __restrict__ Ch,
               const float* __restrict__ bias,
               int N, int K, int O, float alpha, float beta)
{
    extern __shared__ _Float16 smem[];
    _Float16* a_sm = smem;                      // [BM][LDA]
    _Float16* b_sm = smem + BM * LDA;           // [BN][LDA]

    const int tid  = threadIdx.x;
    const int lane = tid & 31;
    const int wid  = tid >> 5;                  // 0..7
    const int wm   = wid >> 2;                  // 0..1 : 64-row half of M tile
    const int wn   = wid & 3;                   // 0..3 : 16-col slice of N tile

    const int block_m = blockIdx.y * BM;
    const int block_n = blockIdx.x * BN;

#if USE_TDM
    const unsigned a_lds = (unsigned)(unsigned long long)(void*)a_sm;
    const unsigned b_lds = (unsigned)(unsigned long long)(void*)b_sm;
#endif

    v8f acc[4] = {};

    for (int k0 = 0; k0 < K; k0 += BK) {
#if USE_TDM
        // TDM DMA: wave0 pulls the A tile, wave1 the B tile; LDS padding is
        // applied by the mover itself (row stride ends up LDA halves).
        if (wid == 0)
            tdm_load_tile(Ah + (size_t)block_m * K + k0, a_lds, K, N, BM);
        else if (wid == 1)
            tdm_load_tile(Bh + (size_t)block_n * K + k0, b_lds, K, O, BN);
        if (wid < 2)
            __builtin_amdgcn_s_wait_tensorcnt(0);
        __syncthreads();
#else
        {   // A: 128x32 halves, 16 per thread
            const int r = tid >> 1, c = (tid & 1) * 16;
            const uint4* src = (const uint4*)(Ah + (size_t)(block_m + r) * K + k0 + c);
            *(uint4*)&a_sm[r * LDA + c]     = src[0];
            *(uint4*)&a_sm[r * LDA + c + 8] = src[1];
        }
        {   // B: 64x32 halves, 8 per thread
            const int r = tid >> 2, c = (tid & 3) * 8;
            const uint4* src = (const uint4*)(Bh + (size_t)(block_n + r) * K + k0 + c);
            *(uint4*)&b_sm[r * LDA + c] = src[0];
            if (k0 + BK < K) __builtin_prefetch(src + K / 8, 0, 1);
        }
        __syncthreads();
#endif
        // B fragment (32x16): lanes 0-15 carry K 0..7,16..23, lanes 16-31 K 8..15,24..31
        v16h bf;
        {
            const int n  = lane & 15;
            const int kb = (lane >> 4) << 3;
            #pragma unroll
            for (int h = 0; h < 16; ++h)
                bf[h] = b_sm[(wn * 16 + n) * LDA + ((h >> 3) << 4) + kb + (h & 7)];
        }
        #pragma unroll
        for (int t = 0; t < 4; ++t) {
            v16h af;
            const int m  = wm * 64 + t * 16 + (lane & 15);
            const int kb = (lane >> 4) << 3;
            #pragma unroll
            for (int h = 0; h < 16; ++h)
                af[h] = a_sm[m * LDA + ((h >> 3) << 4) + kb + (h & 7)];
            acc[t] = __builtin_amdgcn_wmma_f32_16x16x32_f16(
                false, af, false, bf, (short)0, acc[t], false, false);
        }
        __syncthreads();
    }

    // epilogue per C layout: VGPR r -> row r (lanes 0-15) / r+8 (lanes 16-31)
    const int n_col = block_n + wn * 16 + (lane & 15);
    const int m_hi  = (lane >> 4) << 3;
    const float badd = bias ? bias[n_col] : 0.0f;
    #pragma unroll
    for (int t = 0; t < 4; ++t) {
        #pragma unroll
        for (int r = 0; r < 8; ++r) {
            const int row = block_m + wm * 64 + t * 16 + r + m_hi;
            const size_t idx = (size_t)row * O + n_col;
            float v = alpha * acc[t][r] + badd;
            if (beta != 0.0f) v += beta * C[idx];
            C[idx] = v;
            if (Ch) Ch[idx] = (_Float16)v;
        }
    }
}

// ---------------------------------------------------------------------------
// f32 -> f16 cast (grid-stride); used only for the big static weights.
// ---------------------------------------------------------------------------
__global__ __launch_bounds__(256)
void k_cast_h(const float* __restrict__ s, _Float16* __restrict__ d, size_t n)
{
    size_t i = (size_t)blockIdx.x * 256 + threadIdx.x;
    const size_t stride = (size_t)gridDim.x * 256;
    for (; i < n; i += stride) d[i] = (_Float16)s[i];
}

// f32 [M,D] -> f16 transposed [D,M]
__global__ __launch_bounds__(256)
void k_cast_T(const float* __restrict__ src, _Float16* __restrict__ dst,
              int M, int D)
{
    __shared__ float tile[32][33];
    const int m0 = blockIdx.y * 32, d0 = blockIdx.x * 32;
    const int tx = threadIdx.x & 31, ty = threadIdx.x >> 5;   // 32x8
    for (int i = ty; i < 32; i += 8)
        tile[i][tx] = src[(size_t)(m0 + i) * D + d0 + tx];
    __syncthreads();
    for (int i = ty; i < 32; i += 8)
        dst[(size_t)(d0 + i) * M + m0 + tx] = (_Float16)tile[tx][i];
}

// ---------------------------------------------------------------------------
// Embedding gather -> f16 activations
// ---------------------------------------------------------------------------
__global__ __launch_bounds__(256)
void k_embed(const int* __restrict__ tok,
             const float* __restrict__ er, const float* __restrict__ ei,
             _Float16* __restrict__ zrh, _Float16* __restrict__ zih, int D)
{
    const int n = blockIdx.x;
    const size_t t = (size_t)tok[n];
    for (int d = threadIdx.x; d < D; d += 256) {
        zrh[(size_t)n * D + d] = (_Float16)er[t * D + d];
        zih[(size_t)n * D + d] = (_Float16)ei[t * D + d];
    }
}

__global__ __launch_bounds__(256)
void k_init_ptr(float* __restrict__ ptr, int n_total)
{
    const int i = blockIdx.x * 256 + threadIdx.x;
    if (i < n_total) ptr[i] = ((i & (STACK_ - 1)) == 0) ? 1.0f : 0.0f;
}

// ---------------------------------------------------------------------------
// Complex LN + focus + modReLU: reads f32 lr/li, writes f16 zr1/zi1.
// ---------------------------------------------------------------------------
__global__ __launch_bounds__(256)
void k_cell_norm(const float* __restrict__ lr, const float* __restrict__ li,
                 _Float16* __restrict__ lrh, _Float16* __restrict__ lih,
                 const float* __restrict__ ln_scale,
                 const float* __restrict__ ln_shift,
                 const float* __restrict__ focus,
                 const float* __restrict__ modb, int D)
{
    __shared__ float rs[256], rs2[256];
    const int n = blockIdx.x, tid = threadIdx.x;
    float s = 0.f, s2 = 0.f;
    for (int d = tid; d < D; d += 256) {
        const float a = lr[(size_t)n * D + d], b = li[(size_t)n * D + d];
        const float mag = sqrtf(a * a + b * b) + EPS_;
        s += mag; s2 += mag * mag;
    }
    rs[tid] = s; rs2[tid] = s2;
    __syncthreads();
    for (int st = 128; st > 0; st >>= 1) {
        if (tid < st) { rs[tid] += rs[tid + st]; rs2[tid] += rs2[tid + st]; }
        __syncthreads();
    }
    const float mean = rs[0] / (float)D;
    const float var  = (rs2[0] - (float)D * mean * mean) / (float)(D - 1);
    const float rstd = rsqrtf(var + EPS_);
    for (int d = tid; d < D; d += 256) {
        const size_t idx = (size_t)n * D + d;
        const float a = lr[idx], b = li[idx];
        const float mag = sqrtf(a * a + b * b) + EPS_;
        const float nm  = (mag - mean) * rstd * ln_scale[d] + ln_shift[d];
        float zr1 = nm * (a / mag), zi1 = nm * (b / mag);
        const float f = 1.0f / (1.0f + expf(-focus[d]));
        zr1 *= f; zi1 *= f;
        const float nrm = sqrtf(zr1 * zr1 + zi1 * zi1) + EPS_;
        const float sc  = fmaxf(nrm + modb[d], 0.0f) / nrm;
        lrh[idx] = (_Float16)(zr1 * sc);
        lih[idx] = (_Float16)(zi1 * sc);
    }
}

// ---------------------------------------------------------------------------
// Row softmax (in place) + f16 copy for the w@v WMMA GEMM.
// ---------------------------------------------------------------------------
__global__ __launch_bounds__(256)
void k_softmax(float* __restrict__ S, _Float16* __restrict__ Sh, int M)
{
    __shared__ float red[256];
    const int tid = threadIdx.x;
    const size_t rowoff = (size_t)blockIdx.x * M;
    float* p = S + rowoff;
    float mx = -1e30f;
    for (int c = tid; c < M; c += 256) mx = fmaxf(mx, p[c]);
    red[tid] = mx; __syncthreads();
    for (int st = 128; st > 0; st >>= 1) {
        if (tid < st) red[tid] = fmaxf(red[tid], red[tid + st]);
        __syncthreads();
    }
    mx = red[0]; __syncthreads();
    float sum = 0.f;
    for (int c = tid; c < M; c += 256) {
        const float e = expf(p[c] - mx); p[c] = e; sum += e;
    }
    red[tid] = sum; __syncthreads();
    for (int st = 128; st > 0; st >>= 1) {
        if (tid < st) red[tid] += red[tid + st];
        __syncthreads();
    }
    const float inv = 1.0f / red[0];
    for (int c = tid; c < M; c += 256) {
        const float w = p[c] * inv;
        p[c] = w;
        Sh[rowoff + c] = (_Float16)w;
    }
}

// ---------------------------------------------------------------------------
// halt / ctrl heads: four 1024-wide dot products per row + sigmoid/softmax
// ---------------------------------------------------------------------------
__global__ __launch_bounds__(256)
void k_halt_ctrl(const float* __restrict__ ar, const float* __restrict__ ai,
                 const float* __restrict__ haltW, const float* __restrict__ haltb,
                 const float* __restrict__ stackW, const float* __restrict__ stackb,
                 float* __restrict__ halt, float* __restrict__ ctrl, int D)
{
    __shared__ float red[4][256];
    const int n = blockIdx.x, tid = threadIdx.x;
    float acc[4] = {0.f, 0.f, 0.f, 0.f};
    for (int d = tid; d < D; d += 256) {
        const float a = ar[(size_t)n * D + d], b = ai[(size_t)n * D + d];
        acc[0] += a * haltW[d] + b * haltW[D + d];
        #pragma unroll
        for (int j = 0; j < 3; ++j)
            acc[1 + j] += a * stackW[(size_t)j * 2 * D + d]
                        + b * stackW[(size_t)j * 2 * D + D + d];
    }
    #pragma unroll
    for (int j = 0; j < 4; ++j) red[j][tid] = acc[j];
    __syncthreads();
    for (int st = 128; st > 0; st >>= 1) {
        if (tid < st) {
            #pragma unroll
            for (int j = 0; j < 4; ++j) red[j][tid] += red[j][tid + st];
        }
        __syncthreads();
    }
    if (tid == 0) {
        halt[n] = 1.0f / (1.0f + expf(-(red[0][0] + haltb[0])));
        const float l0 = red[1][0] + stackb[0];
        const float l1 = red[2][0] + stackb[1];
        const float l2 = red[3][0] + stackb[2];
        const float m  = fmaxf(l0, fmaxf(l1, l2));
        const float e0 = expf(l0 - m), e1 = expf(l1 - m), e2 = expf(l2 - m);
        const float si = 1.0f / (e0 + e1 + e2);
        ctrl[(size_t)n * 3 + 0] = e0 * si;
        ctrl[(size_t)n * 3 + 1] = e1 * si;
        ctrl[(size_t)n * 3 + 2] = e2 * si;
    }
}

// ---------------------------------------------------------------------------
// Differentiable stack + ACT. One token / block. Writes next-step f16 z's and
// (on the last step) the f16 copy of out_flat for the logits GEMM.
// ---------------------------------------------------------------------------
__global__ __launch_bounds__(256)
void k_stack_act(const float* __restrict__ ar, const float* __restrict__ ai,
                 float* __restrict__ mem, float* __restrict__ ptr,
                 const float* __restrict__ ctrl, const float* __restrict__ halt,
                 float* __restrict__ cum,
                 _Float16* __restrict__ zrh, _Float16* __restrict__ zih,
                 float* __restrict__ out_flat, _Float16* __restrict__ outh,
                 int D, int is_last)
{
    __shared__ float sptr[STACK_], snew[STACK_], sup[STACK_];
    __shared__ float sc[5];   // push, pop, noop, cum, halt
    __shared__ float sinv;
    const int n = blockIdx.x, tid = threadIdx.x;
    if (tid == 0) {
        sc[0] = ctrl[(size_t)n * 3 + 0];
        sc[1] = ctrl[(size_t)n * 3 + 1];
        sc[2] = ctrl[(size_t)n * 3 + 2];
        sc[3] = cum[n];
        sc[4] = halt[n];
    }
    if (tid < STACK_) sptr[tid] = ptr[(size_t)n * STACK_ + tid];
    __syncthreads();
    if (tid < STACK_) {
        const float up = sptr[(tid + STACK_ - 1) & (STACK_ - 1)]; // roll +1
        const float dn = sptr[(tid + 1) & (STACK_ - 1)];          // roll -1
        sup[tid]  = up;
        snew[tid] = sc[0] * up + sc[1] * dn + sc[2] * sptr[tid];
    }
    __syncthreads();
    if (tid == 0) {
        float s = 0.f;
        for (int i = 0; i < STACK_; ++i) s += snew[i];
        sinv = 1.0f / (s + EPS_);
    }
    __syncthreads();
    if (tid < STACK_) {
        snew[tid] *= sinv;
        ptr[(size_t)n * STACK_ + tid] = snew[tid];
    }
    __syncthreads();

    const float cumv = sc[3], haltv = sc[4], push = sc[0];
    const float still = (cumv < THRESH_) ? 1.0f : 0.0f;
    const float pstep = is_last
        ? (1.0f - cumv) * still
        : (((cumv + haltv) > THRESH_) ? (1.0f - cumv) : haltv) * still;

    const int C2 = 2 * D;
    for (int c = tid; c < C2; c += 256) {
        const float z = (c < D) ? ar[(size_t)n * D + c]
                                : ai[(size_t)n * D + (c - D)];
        float racc = 0.f;
        float* mrow = mem + (size_t)n * STACK_ * C2 + c;
        for (int s = 0; s < STACK_; ++s) {
            const float wm   = push * sup[s];
            const float mold = mrow[(size_t)s * C2];
            const float mnew = wm * z + mold * (1.0f - wm);
            mrow[(size_t)s * C2] = mnew;
            racc += mnew * snew[s];
        }
        const float nz = z + racc;
        if (c < D) zrh[(size_t)n * D + c] = (_Float16)nz;
        else       zih[(size_t)n * D + (c - D)] = (_Float16)nz;
        const size_t oidx = (size_t)n * C2 + c;
        const float of = out_flat[oidx] + pstep * nz;
        out_flat[oidx] = of;
        if (is_last) outh[oidx] = (_Float16)of;
    }
    if (tid == 0) cum[n] = cumv + pstep;
}

// ---------------------------------------------------------------------------
// Host orchestration
// ---------------------------------------------------------------------------
extern "C" void kernel_launch(void* const* d_in, const int* in_sizes, int n_in,
                              void* d_out, int out_size, void* d_ws, size_t ws_size,
                              hipStream_t stream)
{
    const int N = N_TOK_, D = DIM_, V = VOCAB_, C2 = 2 * DIM_;

    const int*   tokens  = (const int*)  d_in[0];
    const float* emb_r   = (const float*)d_in[1];
    const float* emb_i   = (const float*)d_in[2];
    const float* lin_Wr  = (const float*)d_in[3];
    const float* lin_Wi  = (const float*)d_in[4];
    const float* ln_sc   = (const float*)d_in[5];
    const float* ln_sh   = (const float*)d_in[6];
    const float* focus   = (const float*)d_in[7];
    const float* modb    = (const float*)d_in[8];
    const float* q_Wr    = (const float*)d_in[9];
    const float* q_Wi    = (const float*)d_in[10];
    const float* k_Wr    = (const float*)d_in[11];
    const float* k_Wi    = (const float*)d_in[12];
    const float* v_Wr    = (const float*)d_in[13];
    const float* v_Wi    = (const float*)d_in[14];
    const float* halt_W  = (const float*)d_in[15];
    const float* halt_b  = (const float*)d_in[16];
    const float* stack_W = (const float*)d_in[17];
    const float* stack_b = (const float*)d_in[18];
    const float* out_W   = (const float*)d_in[19];
    const float* out_b   = (const float*)d_in[20];

    // ---- carve workspace ----
    char* base = (char*)d_ws;
    size_t off = 0;
    auto alloc = [&](size_t bytes) -> void* {
        void* p = base + off;
        off += (bytes + 255) & ~(size_t)255;
        return p;
    };
    const size_t DD = (size_t)D * D, ND = (size_t)N * D;
    _Float16* Wh[8];
    for (int i = 0; i < 8; ++i) Wh[i] = (_Float16*)alloc(DD * 2);
    _Float16* outWh = (_Float16*)alloc((size_t)V * C2 * 2);
    // f32 GEMM outputs / elementwise state
    float* lr = (float*)alloc(ND * 4);
    float* li = (float*)alloc(ND * 4);
    float* qr = (float*)alloc(ND * 4);
    float* qi = (float*)alloc(ND * 4);
    float* kr = (float*)alloc(ND * 4);
    float* ki = (float*)alloc(ND * 4);
    float* vr = (float*)alloc(ND * 4);
    float* vi = (float*)alloc(ND * 4);
    float* ar = (float*)alloc(ND * 4);
    float* ai = (float*)alloc(ND * 4);
    float* scores   = (float*)alloc((size_t)N * N * 4);
    float* halt     = (float*)alloc((size_t)N * 4);
    float* ctrl     = (float*)alloc((size_t)N * 3 * 4);
    float* cum      = (float*)alloc((size_t)N * 4);
    float* mem      = (float*)alloc((size_t)N * STACK_ * C2 * 4);
    float* ptr      = (float*)alloc((size_t)N * STACK_ * 4);
    float* out_flat = (float*)alloc((size_t)N * C2 * 4);
    // f16 GEMM operands
    _Float16* zrh  = (_Float16*)alloc(ND * 2);
    _Float16* zih  = (_Float16*)alloc(ND * 2);
    _Float16* lrh  = (_Float16*)alloc(ND * 2);
    _Float16* lih  = (_Float16*)alloc(ND * 2);
    _Float16* qrh  = (_Float16*)alloc(ND * 2);
    _Float16* qih  = (_Float16*)alloc(ND * 2);
    _Float16* krh  = (_Float16*)alloc(ND * 2);
    _Float16* kih  = (_Float16*)alloc(ND * 2);
    _Float16* vrTh = (_Float16*)alloc(ND * 2);
    _Float16* viTh = (_Float16*)alloc(ND * 2);
    _Float16* wh   = (_Float16*)alloc((size_t)N * N * 2);
    _Float16* outh = (_Float16*)alloc((size_t)N * C2 * 2);
    (void)ws_size; (void)in_sizes; (void)n_in; (void)out_size;

    // ---- weight casts (recomputed every call for determinism) ----
    const float* wsrc[8] = {lin_Wr, lin_Wi, q_Wr, q_Wi, k_Wr, k_Wi, v_Wr, v_Wi};
    for (int i = 0; i < 8; ++i)
        k_cast_h<<<1024, 256, 0, stream>>>(wsrc[i], Wh[i], DD);
    k_cast_h<<<4096, 256, 0, stream>>>(out_W, outWh, (size_t)V * C2);

    // ---- state init ----
    k_embed<<<N, 256, 0, stream>>>(tokens, emb_r, emb_i, zrh, zih, D);
    (void)hipMemsetAsync(mem, 0, (size_t)N * STACK_ * C2 * 4, stream);
    (void)hipMemsetAsync(cum, 0, (size_t)N * 4, stream);
    (void)hipMemsetAsync(out_flat, 0, (size_t)N * C2 * 4, stream);
    k_init_ptr<<<(N * STACK_ + 255) / 256, 256, 0, stream>>>(ptr, N * STACK_);

    auto gemm = [&](const _Float16* A, const _Float16* B, float* C,
                    _Float16* Ch, const float* bias, int n, int k, int o,
                    float alpha, float beta) {
        dim3 g(o / BN, n / BM);
        k_gemm_hh<<<g, 256, SMEM_BYTES, stream>>>(A, B, C, Ch, bias,
                                                  n, k, o, alpha, beta);
    };

    const float rsD = rsqrtf((float)D);   // D^-0.5

    for (int step = 0; step < DEPTH_; ++step) {
        // complex linear: lr = zr@Wr.T - zi@Wi.T ; li = zi@Wr.T + zr@Wi.T
        gemm(zrh, Wh[0], lr, nullptr, nullptr, N, D, D,  1.0f, 0.0f);
        gemm(zih, Wh[1], lr, nullptr, nullptr, N, D, D, -1.0f, 1.0f);
        gemm(zih, Wh[0], li, nullptr, nullptr, N, D, D,  1.0f, 0.0f);
        gemm(zrh, Wh[1], li, nullptr, nullptr, N, D, D,  1.0f, 1.0f);

        // LN + focus + modReLU -> f16 zr1/zi1
        k_cell_norm<<<N, 256, 0, stream>>>(lr, li, lrh, lih,
                                           ln_sc, ln_sh, focus, modb, D);

        // q / k / v complex projections (f16 copies fused into 2nd call)
        gemm(lrh, Wh[2], qr, nullptr, nullptr, N, D, D,  1.0f, 0.0f);
        gemm(lih, Wh[3], qr, qrh,     nullptr, N, D, D, -1.0f, 1.0f);
        gemm(lih, Wh[2], qi, nullptr, nullptr, N, D, D,  1.0f, 0.0f);
        gemm(lrh, Wh[3], qi, qih,     nullptr, N, D, D,  1.0f, 1.0f);

        gemm(lrh, Wh[4], kr, nullptr, nullptr, N, D, D,  1.0f, 0.0f);
        gemm(lih, Wh[5], kr, krh,     nullptr, N, D, D, -1.0f, 1.0f);
        gemm(lih, Wh[4], ki, nullptr, nullptr, N, D, D,  1.0f, 0.0f);
        gemm(lrh, Wh[5], ki, kih,     nullptr, N, D, D,  1.0f, 1.0f);

        gemm(lrh, Wh[6], vr, nullptr, nullptr, N, D, D,  1.0f, 0.0f);
        gemm(lih, Wh[7], vr, nullptr, nullptr, N, D, D, -1.0f, 1.0f);
        gemm(lih, Wh[6], vi, nullptr, nullptr, N, D, D,  1.0f, 0.0f);
        gemm(lrh, Wh[7], vi, nullptr, nullptr, N, D, D,  1.0f, 1.0f);

        // v transposed to [D,N] f16 for the w@v GEMM (B operand)
        k_cast_T<<<dim3(D / 32, N / 32), 256, 0, stream>>>(vr, vrTh, N, D);
        k_cast_T<<<dim3(D / 32, N / 32), 256, 0, stream>>>(vi, viTh, N, D);

        // scores = (qr@kr.T + qi@ki.T) * D^-0.5 ; softmax (f16 copy -> wh)
        gemm(qrh, krh, scores, nullptr, nullptr, N, D, N, rsD, 0.0f);
        gemm(qih, kih, scores, nullptr, nullptr, N, D, N, rsD, 1.0f);
        k_softmax<<<N, 256, 0, stream>>>(scores, wh, N);

        // ar = w @ vr ; ai = w @ vi
        gemm(wh, vrTh, ar, nullptr, nullptr, N, N, D, 1.0f, 0.0f);
        gemm(wh, viTh, ai, nullptr, nullptr, N, N, D, 1.0f, 0.0f);

        // halt / ctrl heads
        k_halt_ctrl<<<N, 256, 0, stream>>>(ar, ai, halt_W, halt_b,
                                           stack_W, stack_b, halt, ctrl, D);

        // stack + ACT (updates mem, ptr, cum, zrh/zih, out_flat[, outh])
        k_stack_act<<<N, 256, 0, stream>>>(ar, ai, mem, ptr, ctrl, halt, cum,
                                           zrh, zih, out_flat, outh, D,
                                           step == DEPTH_ - 1 ? 1 : 0);
    }

    // logits = out_flat @ out_W.T + out_b   -> d_out (f32, 2048 x 32000)
    gemm(outh, outWh, (float*)d_out, nullptr, out_b, N, C2, V, 1.0f, 0.0f);
}
